// PointerNetwork_69707319214358
// MI455X (gfx1250) — matrix operand order repaired and checked by reading, not compiled, not run on hardware
//
#include <hip/hip_runtime.h>
#include <hip/hip_bf16.h>
#include <climits>

#define B_   256
#define N_   512
#define E_   256
#define K_   256
#define H_   256
#define IN_  1024
#define F_   1024
#define S_   64
#define G4H  1024   // 4*H

typedef __attribute__((ext_vector_type(16))) _Float16 v16h;
typedef __attribute__((ext_vector_type(8)))  float    v8f;

union FragH { v16h v; uint4 q[2]; };

// ---------------------------------------------------------------- conversions
__global__ void k_cvt(const float* __restrict__ src, _Float16* __restrict__ dst, size_t n) {
  size_t i = (size_t)blockIdx.x * blockDim.x + threadIdx.x;
  if (i < n) dst[i] = (_Float16)src[i];
}
// src [R,C] row-major -> dst [C,R] row-major (i.e. transpose) as f16
__global__ void k_cvtT(const float* __restrict__ src, _Float16* __restrict__ dst, int R, int C) {
  size_t i = (size_t)blockIdx.x * blockDim.x + threadIdx.x;
  if (i < (size_t)R * C) {
    int r = (int)(i / C), c = (int)(i % C);
    dst[(size_t)c * R + r] = (_Float16)src[i];
  }
}
__global__ void k_copy_f32(const float* __restrict__ src, float* __restrict__ dst, size_t n) {
  size_t i = (size_t)blockIdx.x * blockDim.x + threadIdx.x;
  if (i < n) dst[i] = src[i];
}

// ---------------------------------------------------------------- WMMA fragment helpers
__device__ inline void load_a_frag(FragH& a, const _Float16* __restrict__ arow, int k0, int hi) {
  const _Float16* pa = arow + k0 + hi * 8;     // K: base..base+7, base+16..base+23
  a.q[0] = *(const uint4*)(pa);
  a.q[1] = *(const uint4*)(pa + 16);
}
__device__ inline void load_b_frag(FragH& b, const _Float16* __restrict__ BT, int ldb,
                                   int col, int k0, int hi) {
  const _Float16* pb = BT + (size_t)col * ldb + k0 + hi * 8;   // K: base..base+7, base+16..base+23
  b.q[0] = *(const uint4*)(pb);
  b.q[1] = *(const uint4*)(pb + 16);
}

// Double-buffered 16x64 output tile per wave; requires Kd % 64 == 0.
__device__ inline void gemm_tile_16x64(v8f acc[4], const _Float16* __restrict__ arow,
                                       const _Float16* __restrict__ BT, int ldb,
                                       int ncol, int l16, int hi, int Kd)
{
  FragH aA, bA[4], aB, bB[4];
  load_a_frag(aA, arow, 0, hi);
  #pragma unroll
  for (int j = 0; j < 4; ++j) load_b_frag(bA[j], BT, ldb, ncol + j * 16 + l16, 0, hi);

  for (int k0 = 0; k0 < Kd; k0 += 64) {
    // prefetch second half-slice while first multiplies
    load_a_frag(aB, arow, k0 + 32, hi);
    #pragma unroll
    for (int j = 0; j < 4; ++j) load_b_frag(bB[j], BT, ldb, ncol + j * 16 + l16, k0 + 32, hi);
    #pragma unroll
    for (int j = 0; j < 4; ++j)
      acc[j] = __builtin_amdgcn_wmma_f32_16x16x32_f16(false, aA.v, false, bA[j].v,
                                                      (short)0, acc[j], false, false);
    if (k0 + 64 < Kd) {
      load_a_frag(aA, arow, k0 + 64, hi);
      #pragma unroll
      for (int j = 0; j < 4; ++j) load_b_frag(bA[j], BT, ldb, ncol + j * 16 + l16, k0 + 64, hi);
    }
    #pragma unroll
    for (int j = 0; j < 4; ++j)
      acc[j] = __builtin_amdgcn_wmma_f32_16x16x32_f16(false, aB.v, false, bB[j].v,
                                                      (short)0, acc[j], false, false);
  }
}

// ---------------------------------------------------------------- generic WMMA GEMM
// C[M,N] = A[M,Kd] (f16 row-major) @ BT[N,Kd]^T (f16, B pre-transposed)
// flags: 1=+bias[N], 2=relu, 4=+resid[(row%resMod)*N+col], 8=write f32, 16=write f16
__global__ __launch_bounds__(256) void k_gemm(
    const _Float16* __restrict__ A, const _Float16* __restrict__ BT,
    float* __restrict__ Cf, _Float16* __restrict__ Ch,
    const float* __restrict__ bias, const float* __restrict__ resid,
    int M, int N, int Kd, int resMod, int flags)
{
  const int w    = threadIdx.x >> 5;
  const int lane = threadIdx.x & 31;
  const int l16  = lane & 15;
  const int hi   = lane >> 4;
  const int mrow = blockIdx.x * 128 + w * 16;   // 8 waves: 128 rows per WG
  const int ncol = blockIdx.y * 64;             // 64 cols per wave (4 tiles)
  if (mrow >= M) return;

  v8f acc[4];
  #pragma unroll
  for (int j = 0; j < 4; ++j)
    #pragma unroll
    for (int e = 0; e < 8; ++e) acc[j][e] = 0.0f;

  const _Float16* arow = A + (size_t)(mrow + l16) * Kd;
  gemm_tile_16x64(acc, arow, BT, Kd, ncol, l16, hi, Kd);

  #pragma unroll
  for (int j = 0; j < 4; ++j) {
    #pragma unroll
    for (int e = 0; e < 8; ++e) {
      int r = mrow + hi * 8 + e;
      int c = ncol + j * 16 + l16;
      float v = acc[j][e];
      if (flags & 1) v += bias[c];
      if (flags & 2) v = fmaxf(v, 0.0f);
      if (flags & 4) v += resid[(size_t)(r % resMod) * N + c];
      size_t o = (size_t)r * N + c;
      if (flags & 8)  Cf[o] = v;
      if (flags & 16) Ch[o] = (_Float16)v;
    }
  }
}

// ---------------------------------------------------------------- selection bookkeeping
__global__ void k_first(const int* __restrict__ sel, int* __restrict__ flg) {
  int b = blockIdx.x, s = threadIdx.x;
  int v = sel[b * S_ + s], f = 1;
  for (int j = 0; j < s; ++j) if (sel[b * S_ + j] == v) { f = 0; break; }
  flg[b * S_ + s] = f;
}
__global__ void k_selstep_init(int* __restrict__ ss) {
  size_t i = (size_t)blockIdx.x * blockDim.x + threadIdx.x;
  if (i < (size_t)B_ * N_) ss[i] = INT_MAX;
}
__global__ void k_selstep_fill(const int* __restrict__ sel, int* __restrict__ ss) {
  int b = blockIdx.x, s = threadIdx.x;
  atomicMin(&ss[b * N_ + sel[b * S_ + s]], s);
}
// su[(t-1)*B + b][k] = (sum of first t deduped selected key rows)/t, as f16
__global__ void k_su(const float* __restrict__ key, const int* __restrict__ sel,
                     const int* __restrict__ flg, _Float16* __restrict__ su) {
  int b = blockIdx.x, k = threadIdx.x;
  float acc = 0.0f;
  for (int t = 1; t <= S_; ++t) {
    int s = t - 1;
    if (flg[b * S_ + s]) acc += key[((size_t)b * N_ + sel[b * S_ + s]) * K_ + k];
    su[((size_t)(t - 1) * B_ + b) * K_ + k] = (_Float16)(acc / (float)t);
  }
}
// aeq[s*B+b][i] = (s==0 ? embedding : ae_out[s-1]) as f16
__global__ void k_aeq(const float* __restrict__ emb, const float* __restrict__ aeo,
                      _Float16* __restrict__ aeq) {
  size_t i = (size_t)blockIdx.x * blockDim.x + threadIdx.x;
  if (i >= (size_t)S_ * B_ * IN_) return;
  int s = (int)(i / ((size_t)B_ * IN_));
  size_t rem = i % ((size_t)B_ * IN_);
  int b = (int)(rem / IN_), ii = (int)(rem % IN_);
  float v = (s == 0) ? emb[(size_t)b * IN_ + ii]
                     : aeo[((size_t)(s - 1) * B_ + b) * IN_ + ii];
  aeq[i] = (_Float16)v;
}

// ---------------------------------------------------------------- LayerNorm-LSTM cell
__device__ inline float blk_sum(float v, float* red) {
  red[threadIdx.x] = v; __syncthreads();
  for (int s = 128; s > 0; s >>= 1) {
    if ((int)threadIdx.x < s) red[threadIdx.x] += red[threadIdx.x + s];
    __syncthreads();
  }
  float r = red[0]; __syncthreads();
  return r;
}
__device__ inline float sigm(float x) { return 1.0f / (1.0f + expf(-x)); }

// one workgroup (H_=256 threads) per batch row; gates = LN(xa)+LN(xb); LN on cell.
__global__ __launch_bounds__(256) void k_cell(
    const float* __restrict__ xa, const float* __restrict__ lag, const float* __restrict__ lab,
    const float* __restrict__ xb, const float* __restrict__ lbg, const float* __restrict__ lbb,
    float* __restrict__ c, const float* __restrict__ lcg, const float* __restrict__ lcb,
    _Float16* __restrict__ hOut, _Float16* __restrict__ qOut, size_t qStride)
{
  __shared__ float red[256];
  int b = blockIdx.x, t = threadIdx.x;
  const float* xar = xa + (size_t)b * G4H;
  const float* xbr = xb + (size_t)b * G4H;
  float av[4], bv[4];
  #pragma unroll
  for (int g = 0; g < 4; ++g) { av[g] = xar[t + g * H_]; bv[g] = xbr[t + g * H_]; }
  float sa = 0, sa2 = 0, sb = 0, sb2 = 0;
  #pragma unroll
  for (int g = 0; g < 4; ++g) { sa += av[g]; sa2 += av[g] * av[g]; sb += bv[g]; sb2 += bv[g] * bv[g]; }
  sa = blk_sum(sa, red); sa2 = blk_sum(sa2, red);
  sb = blk_sum(sb, red); sb2 = blk_sum(sb2, red);
  float mua = sa / (float)G4H, rsa = rsqrtf(sa2 / (float)G4H - mua * mua + 1e-5f);
  float mub = sb / (float)G4H, rsb = rsqrtf(sb2 / (float)G4H - mub * mub + 1e-5f);
  float gate[4];
  #pragma unroll
  for (int g = 0; g < 4; ++g) {
    int idx = t + g * H_;
    gate[g] = (av[g] - mua) * rsa * lag[idx] + lab[idx]
            + (bv[g] - mub) * rsb * lbg[idx] + lbb[idx];
  }
  float cold = c[(size_t)b * H_ + t];
  float cn = sigm(gate[1]) * cold + sigm(gate[0]) * tanhf(gate[2]);
  float sc = blk_sum(cn, red), sc2 = blk_sum(cn * cn, red);
  float muc = sc / (float)H_, rsc = rsqrtf(sc2 / (float)H_ - muc * muc + 1e-5f);
  float cln = (cn - muc) * rsc * lcg[t] + lcb[t];
  float h = sigm(gate[3]) * tanhf(cln);
  c[(size_t)b * H_ + t] = cln;
  hOut[(size_t)b * H_ + t] = (_Float16)h;
  if (qOut) qOut[(size_t)b * qStride + t] = (_Float16)h;
}

// ---------------------------------------------------------------- logits (per-b WMMA GEMM + mask)
__global__ __launch_bounds__(256) void k_logits(
    const _Float16* __restrict__ Q, const _Float16* __restrict__ keyh,
    const unsigned char* __restrict__ emask, const int* __restrict__ selstep,
    float* __restrict__ out)
{
  const int b    = blockIdx.y;
  const int w    = threadIdx.x >> 5;
  const int lane = threadIdx.x & 31;
  const int l16  = lane & 15;
  const int hi   = lane >> 4;
  const int mrow = (w & 3) * 16;                         // s tile (M=64)
  const int ncol = blockIdx.x * 128 + (w >> 2) * 64;     // n tile
  const _Float16* A  = Q    + (size_t)b * S_ * H_;       // [S,H]
  const _Float16* BT = keyh + (size_t)b * N_ * K_;       // key_b [N,K] == (key_b^T)^T

  v8f acc[4];
  #pragma unroll
  for (int j = 0; j < 4; ++j)
    #pragma unroll
    for (int e = 0; e < 8; ++e) acc[j][e] = 0.0f;

  const _Float16* arow = A + (size_t)(mrow + l16) * H_;
  gemm_tile_16x64(acc, arow, BT, K_, ncol, l16, hi, K_);

  #pragma unroll
  for (int j = 0; j < 4; ++j) {
    #pragma unroll
    for (int e = 0; e < 8; ++e) {
      int s = mrow + hi * 8 + e;
      int n = ncol + j * 16 + l16;
      bool ok = (emask[(size_t)b * N_ + n] != 0) && (selstep[(size_t)b * N_ + n] >= s);
      out[((size_t)b * S_ + s) * N_ + n] = ok ? acc[j][e] : -1e9f;
    }
  }
}

// ---------------------------------------------------------------- actor/critic heads
__global__ __launch_bounds__(256) void k_value(
    const float* __restrict__ ae, const float* __restrict__ aw, const float* __restrict__ ab,
    const float* __restrict__ cw, const float* __restrict__ cb,
    float* __restrict__ vl, float* __restrict__ val)
{
  __shared__ float red[256];
  int b = blockIdx.x, t = threadIdx.x;
  float a = 0, c = 0;
  #pragma unroll
  for (int g = 0; g < 4; ++g) {
    int idx = t + g * 256;
    float x = ae[(size_t)b * IN_ + idx];
    a += x * aw[idx]; c += x * cw[idx];
  }
  a = blk_sum(a, red); c = blk_sum(c, red);
  if (t == 0) { vl[b] = a + ab[0]; val[b] = c + cb[0]; }
}

// ---------------------------------------------------------------- host
extern "C" void kernel_launch(void* const* d_in, const int* in_sizes, int n_in,
                              void* d_out, int out_size, void* d_ws, size_t ws_size,
                              hipStream_t stream) {
  (void)in_sizes; (void)n_in; (void)out_size; (void)ws_size;
  const float* emb   = (const float*)d_in[0];
  const float* ent   = (const float*)d_in[1];
  const float* kfw   = (const float*)d_in[2];
  const float* kfb   = (const float*)d_in[3];
  const float* q1w   = (const float*)d_in[4];
  const float* q1b   = (const float*)d_in[5];
  const float* q2w   = (const float*)d_in[6];
  const float* q2b   = (const float*)d_in[7];
  const float* e1w   = (const float*)d_in[8];
  const float* e1b   = (const float*)d_in[9];
  const float* e2w   = (const float*)d_in[10];
  const float* e2b   = (const float*)d_in[11];
  const float* Wih   = (const float*)d_in[12];
  const float* Whh   = (const float*)d_in[13];
  const float* lnihg = (const float*)d_in[14];
  const float* lnihb = (const float*)d_in[15];
  const float* lnhhg = (const float*)d_in[16];
  const float* lnhhb = (const float*)d_in[17];
  const float* lncg  = (const float*)d_in[18];
  const float* lncb  = (const float*)d_in[19];
  const float* aw    = (const float*)d_in[20];
  const float* ab    = (const float*)d_in[21];
  const float* cw    = (const float*)d_in[22];
  const float* cb    = (const float*)d_in[23];
  const unsigned char* emask = (const unsigned char*)d_in[24];
  const int* selu    = (const int*)d_in[25];
  float* out = (float*)d_out;

  char* w = (char*)d_ws;
  size_t off = 0;
  auto carve = [&](size_t bytes) -> char* {
    char* p = w + off; off = (off + bytes + 255) & ~(size_t)255; return p;
  };
  float*    key_f = (float*)   carve((size_t)B_ * N_ * K_ * 4);
  _Float16* key_h = (_Float16*)carve((size_t)B_ * N_ * K_ * 2);
  _Float16* ent_h = (_Float16*)carve((size_t)B_ * N_ * E_ * 2);
  _Float16* kfwT  = (_Float16*)carve((size_t)K_ * E_ * 2);
  _Float16* q1T   = (_Float16*)carve((size_t)F_ * IN_ * 2);
  _Float16* q2T   = (_Float16*)carve((size_t)K_ * F_ * 2);
  _Float16* e1T   = (_Float16*)carve((size_t)F_ * K_ * 2);
  _Float16* e2T   = (_Float16*)carve((size_t)IN_ * F_ * 2);
  _Float16* wih0  = (_Float16*)carve((size_t)G4H * K_ * 2);
  _Float16* whh0  = (_Float16*)carve((size_t)G4H * H_ * 2);
  _Float16* wih1  = (_Float16*)carve((size_t)G4H * H_ * 2);
  _Float16* whh1  = (_Float16*)carve((size_t)G4H * H_ * 2);
  int*      flg   = (int*)     carve((size_t)B_ * S_ * 4);
  int*      sstep = (int*)     carve((size_t)B_ * N_ * 4);
  _Float16* suh   = (_Float16*)carve((size_t)S_ * B_ * K_ * 2);
  _Float16* ehid  = (_Float16*)carve((size_t)S_ * B_ * F_ * 2);
  float*    aeout = (float*)   carve((size_t)S_ * B_ * IN_ * 4);
  _Float16* aeq   = (_Float16*)carve((size_t)S_ * B_ * IN_ * 2);
  _Float16* qhid  = (_Float16*)carve((size_t)S_ * B_ * F_ * 2);
  _Float16* qall  = (_Float16*)carve((size_t)S_ * B_ * K_ * 2);
  float*    xg0   = (float*)   carve((size_t)S_ * B_ * G4H * 4);
  float*    hg0   = (float*)   carve((size_t)B_ * G4H * 4);
  float*    xg1   = (float*)   carve((size_t)B_ * G4H * 4);
  float*    hg1   = (float*)   carve((size_t)B_ * G4H * 4);
  _Float16* h0h   = (_Float16*)carve((size_t)B_ * H_ * 2);
  _Float16* h1h   = (_Float16*)carve((size_t)B_ * H_ * 2);
  float*    c0    = (float*)   carve((size_t)B_ * H_ * 4);
  float*    c1    = (float*)   carve((size_t)B_ * H_ * 4);
  _Float16* Qb    = (_Float16*)carve((size_t)B_ * S_ * H_ * 2);

  hipMemsetAsync(h0h, 0, (size_t)B_ * H_ * 2, stream);
  hipMemsetAsync(h1h, 0, (size_t)B_ * H_ * 2, stream);
  hipMemsetAsync(c0,  0, (size_t)B_ * H_ * 4, stream);
  hipMemsetAsync(c1,  0, (size_t)B_ * H_ * 4, stream);

  const int T = 256;
  auto nb = [&](size_t n) { return (unsigned)((n + T - 1) / T); };
  dim3 blk(256);

  // weight / input conversions
  k_cvt <<<nb((size_t)B_ * N_ * E_), T, 0, stream>>>(ent, ent_h, (size_t)B_ * N_ * E_);
  k_cvtT<<<nb((size_t)E_ * K_),  T, 0, stream>>>(kfw, kfwT, E_, K_);
  k_cvtT<<<nb((size_t)IN_ * F_), T, 0, stream>>>(q1w, q1T, IN_, F_);
  k_cvtT<<<nb((size_t)F_ * K_),  T, 0, stream>>>(q2w, q2T, F_, K_);
  k_cvtT<<<nb((size_t)K_ * F_),  T, 0, stream>>>(e1w, e1T, K_, F_);
  k_cvtT<<<nb((size_t)F_ * IN_), T, 0, stream>>>(e2w, e2T, F_, IN_);
  k_cvt <<<nb((size_t)G4H * K_), T, 0, stream>>>(Wih,             wih0, (size_t)G4H * K_);
  k_cvt <<<nb((size_t)G4H * H_), T, 0, stream>>>(Wih + G4H * K_,  wih1, (size_t)G4H * H_);
  k_cvt <<<nb((size_t)G4H * H_), T, 0, stream>>>(Whh,             whh0, (size_t)G4H * H_);
  k_cvt <<<nb((size_t)G4H * H_), T, 0, stream>>>(Whh + G4H * H_,  whh1, (size_t)G4H * H_);

  // key = ent @ key_fc_w + b  (f32 + f16 copies)
  k_gemm<<<dim3((B_ * N_) / 128, K_ / 64), blk, 0, stream>>>(
      ent_h, kfwT, key_f, key_h, kfb, nullptr, B_ * N_, K_, E_, 1, 1 | 8 | 16);

  // selection bookkeeping
  k_first<<<B_, S_, 0, stream>>>(selu, flg);
  k_selstep_init<<<nb((size_t)B_ * N_), T, 0, stream>>>(sstep);
  k_selstep_fill<<<B_, S_, 0, stream>>>(selu, sstep);
  k_su<<<B_, K_, 0, stream>>>(key_f, selu, flg, suh);

  // ae for all steps: ae_t = base + (relu(su_t@e1+b)@e2+b), t=1..64
  k_gemm<<<dim3(S_ * B_ / 128, F_ / 64), blk, 0, stream>>>(
      suh, e1T, nullptr, ehid, e1b, nullptr, S_ * B_, F_, K_, 1, 1 | 2 | 16);
  k_gemm<<<dim3(S_ * B_ / 128, IN_ / 64), blk, 0, stream>>>(
      ehid, e2T, aeout, nullptr, e2b, emb, S_ * B_, IN_, F_, B_, 1 | 4 | 8);
  k_aeq<<<nb((size_t)S_ * B_ * IN_), T, 0, stream>>>(emb, aeout, aeq);

  // query MLP for all steps + Wih0 projection (batched)
  k_gemm<<<dim3(S_ * B_ / 128, F_ / 64), blk, 0, stream>>>(
      aeq, q1T, nullptr, qhid, q1b, nullptr, S_ * B_, F_, IN_, 1, 1 | 2 | 16);
  k_gemm<<<dim3(S_ * B_ / 128, K_ / 64), blk, 0, stream>>>(
      qhid, q2T, nullptr, qall, q2b, nullptr, S_ * B_, K_, F_, 1, 1 | 16);
  k_gemm<<<dim3(S_ * B_ / 128, G4H / 64), blk, 0, stream>>>(
      qall, wih0, xg0, nullptr, nullptr, nullptr, S_ * B_, G4H, K_, 1, 8);

  // sequential LayerNorm-LSTM (only the h-recurrences)
  for (int s = 0; s < S_; ++s) {
    k_gemm<<<dim3(2, G4H / 64), blk, 0, stream>>>(
        h0h, whh0, hg0, nullptr, nullptr, nullptr, B_, G4H, H_, 1, 8);
    k_cell<<<B_, H_, 0, stream>>>(
        xg0 + (size_t)s * B_ * G4H, lnihg, lnihb, hg0, lnhhg, lnhhb,
        c0, lncg, lncb, h0h, nullptr, 0);
    k_gemm<<<dim3(2, G4H / 64), blk, 0, stream>>>(
        h0h, wih1, xg1, nullptr, nullptr, nullptr, B_, G4H, H_, 1, 8);
    k_gemm<<<dim3(2, G4H / 64), blk, 0, stream>>>(
        h1h, whh1, hg1, nullptr, nullptr, nullptr, B_, G4H, H_, 1, 8);
    k_cell<<<B_, H_, 0, stream>>>(
        xg1, lnihg + G4H, lnihb + G4H, hg1, lnhhg + G4H, lnhhb + G4H,
        c1, lncg + H_, lncb + H_, h1h, Qb + (size_t)s * H_, (size_t)S_ * H_);
  }

  // logits for all (b,s,n) with masking; then ae / value heads
  k_logits<<<dim3(N_ / 128, B_), blk, 0, stream>>>(Qb, key_h, emask, sstep, out);
  const float* aefin = aeout + (size_t)(S_ - 1) * B_ * IN_;
  float* out_ae = out + (size_t)B_ * S_ * N_;
  k_copy_f32<<<nb((size_t)B_ * IN_), T, 0, stream>>>(aefin, out_ae, (size_t)B_ * IN_);
  k_value<<<B_, 256, 0, stream>>>(aefin, aw, ab, cw, cb,
                                  out_ae + (size_t)B_ * IN_,
                                  out_ae + (size_t)B_ * IN_ + B_);
}
